// ConditionalConv_23596550324616
// MI455X (gfx1250) — matrix operand order repaired and checked by reading, not compiled, not run on hardware
//
#include <hip/hip_runtime.h>
#include <cmath>

typedef __attribute__((ext_vector_type(2)))  float        v2f;
typedef __attribute__((ext_vector_type(8)))  float        v8f;
typedef __attribute__((ext_vector_type(16))) __bf16       v16bf;
typedef __attribute__((ext_vector_type(4)))  unsigned int v4u;
typedef __attribute__((ext_vector_type(8)))  int          v8i;
typedef __attribute__((ext_vector_type(4)))  int          v4i;

#define NB     16
#define COND_C 256
#define INC    64
#define OUTC   64
#define HWDIM  128
#define NPAR   36928   // 3*3*64*64 + 64
#define NW     36864   // 3*3*64*64

#if defined(__has_builtin)
#  if __has_builtin(__builtin_amdgcn_tensor_load_to_lds) && \
      __has_builtin(__builtin_amdgcn_s_wait_tensorcnt)
#    define HAVE_TDM 1
#  endif
#endif
#ifndef HAVE_TDM
#  define HAVE_TDM 0
#endif

__device__ __forceinline__ void split_bf16(float x, __bf16& h, __bf16& l) {
    h = (__bf16)x;
    l = (__bf16)(x - (float)h);
}

#if HAVE_TDM
// One TDM descriptor: contiguous 4096 x 2-byte tile, global -> LDS.
// D# group0: [1:0]=count=1, [63:32]=lds_addr, [120:64]=global_addr, [127:126]=type=2
// D# group1: data_size=1(2B); tensor_dim0=tile_dim0=tensor_dim0_stride=4096; dim1=1
// This toolchain exposes the 6-arg builtin:
//   (uint32x4 g0, int32x8 g1, int32x4 g2, int32x4 g3, int32x8 extra, i32 cpol)
__device__ __forceinline__ void tdm_load_chunk(const __bf16* gsrc, unsigned lds_byte_off) {
    unsigned long long ga = (unsigned long long)(size_t)gsrc;
    v4u g0;
    g0.x = 1u;                                            // count=1, user mode
    g0.y = lds_byte_off;                                  // lds_addr
    g0.z = (unsigned)ga;                                  // global_addr[31:0]
    g0.w = (unsigned)((ga >> 32) & 0x1FFFFFFull) | (2u << 30);  // [24:0]=ga[56:32], type=2
    v8i g1 = { 0x10000,                 // data_size=2B (bits 17:16 = 1)
               (int)(4096u << 16),      // tensor_dim0[15:0] @ bits 63:48
               1 << 16,                 // tensor_dim1[15:0]=1 @ bits 95:80
               (int)(4096u << 16),      // tile_dim0=4096 @ bits 127:112
               1,                       // tile_dim1=1 @ bits 143:128
               4096,                    // tensor_dim0_stride[31:0]
               0, 0 };
    v4i z4 = {0, 0, 0, 0};
    v8i z8 = {0, 0, 0, 0, 0, 0, 0, 0};
    __builtin_amdgcn_tensor_load_to_lds(g0, g1, z4, z4, z8, 0);
}
#endif

// ---------------------------------------------------------------------------
// Kernel 1: w_b = tanh(cond @ W_cond^T + b_cond) * 5   via V_WMMA_F32_16X16X4_F32
// M = 16 (batch = one tile), N = 36928, K = 256.  Weights written pre-split
// (bf16 hi/lo) and pre-packed per TDM chunk: [b][p][half][hi 32x64 | lo 32x64].
// ---------------------------------------------------------------------------
__global__ __launch_bounds__(128)
void hyper_gemm(const float* __restrict__ cond, const float* __restrict__ Wc,
                const float* __restrict__ bc,
                __bf16* __restrict__ w_pk, float* __restrict__ bias)
{
    __shared__ float sc[NB * COND_C];            // 16 KB conditioning matrix
    const int tid = threadIdx.x;
    for (int i = tid; i < NB * COND_C; i += 128) sc[i] = cond[i];
    __syncthreads();

    const int wid  = tid >> 5;
    const int lane = tid & 31;
    const int tile = blockIdx.x * 4 + wid;       // 2308 tiles, exact
    const int n0   = tile * 16;
    const int mrow = lane & 15;
    const int hsel = lane >> 4;
    const int koff = hsel * 2;

    const float* wrow = Wc + (size_t)(n0 + mrow) * COND_C;   // B[k][n] = W_cond[n][k]
    const float* crow = sc + mrow * COND_C;                  // A[m][k] = cond[m][k]

    v8f acc = {};
    for (int kk = 0; kk < COND_C; kk += 4) {
        v2f a, b;
        a[0] = crow[kk + koff];     a[1] = crow[kk + koff + 1];
        b[0] = wrow[kk + koff];     b[1] = wrow[kk + koff + 1];
        acc = __builtin_amdgcn_wmma_f32_16x16x4_f32(false, a, false, b,
                                                    (short)0, acc, false, false);
    }

    const int n    = n0 + mrow;
    const float bcv = bc[n];
    #pragma unroll
    for (int r = 0; r < 8; ++r) {
        const int m = r + 8 * hsel;              // batch index
        const float v = tanhf(acc[r] + bcv) * 5.0f;
        if (n < NW) {
            const int oc  = n / 576;
            const int rem = n % 576;
            const int ic  = rem / 9;
            const int p   = rem % 9;             // kh*3+kw
            const int half = ic >> 5, ics = ic & 31;
            const size_t base = ((size_t)(m * 9 + p) * 2 + half) * 4096;
            __bf16 h, l; split_bf16(v, h, l);
            w_pk[base + ics * OUTC + oc]        = h;
            w_pk[base + 2048 + ics * OUTC + oc] = l;
        } else {
            bias[m * OUTC + (n - NW)] = v;
        }
    }
}

// ---------------------------------------------------------------------------
// Kernel 2: per-batch conv as 9 shifted GEMMs, split-bf16 precision.
// WG = 256 thr (8 waves as 4M x 2N), tile = 64 oc x 32 pixels (one image row).
// Weight chunks (8 KB) streamed into LDS by the Tensor Data Mover, double-
// buffered against TENSORcnt; 54 v_wmma_f32_16x16x32_bf16 per wave.
// ---------------------------------------------------------------------------
__global__ __launch_bounds__(256)
void conv_wmma(const float* __restrict__ x,
               const __bf16* __restrict__ w_pk,
               const float* __restrict__ bias, float* __restrict__ out)
{
    constexpr int WSTR = 36;                  // padded col stride (34 used)
    constexpr int WROW = 3 * WSTR;            // 108 per ic
    constexpr int XH   = 0;
    constexpr int XL   = INC * WROW;          // 6912
    constexpr int AOFF = 2 * INC * WROW;      // 13824
    __shared__ __align__(16) __bf16 smem[AOFF + 2 * 4096];   // 44 KB

    const int b   = blockIdx.y;
    const int t   = blockIdx.x;               // 512 pixel tiles per batch
    const int h   = t >> 2;
    const int w0  = (t & 3) * 32;
    const int tid = threadIdx.x;

    // Stage 64ic x 3row x 34col zero-padded input window, hi/lo split on store.
    const float* xb = x + (size_t)b * INC * HWDIM * HWDIM;
    for (int i = tid; i < INC * 3 * 34; i += 256) {
        const int ic = i / 102, rem = i % 102, r = rem / 34, c = rem % 34;
        const int gh = h - 1 + r, gw = w0 - 1 + c;
        float v = 0.0f;
        if ((unsigned)gh < (unsigned)HWDIM && (unsigned)gw < (unsigned)HWDIM)
            v = xb[(size_t)ic * HWDIM * HWDIM + gh * HWDIM + gw];
        __bf16 hh, ll; split_bf16(v, hh, ll);
        smem[XH + ic * WROW + r * WSTR + c] = hh;
        smem[XL + ic * WROW + r * WSTR + c] = ll;
    }

    const int wid  = tid >> 5, lane = tid & 31;
    const int oc0  = (wid & 3) * 16;
    const int nOff = (wid >> 2) * 16;
    const int m    = lane & 15;
    const int hsel = lane >> 4;

    const __bf16* wb = w_pk + (size_t)b * 18 * 4096;
    const unsigned smem_base = (unsigned)(size_t)&smem[0];   // LDS byte offset
    (void)smem_base;

#if HAVE_TDM
    if (wid == 0) {      // one wave drives the TDM; descriptors are uniform
        tdm_load_chunk(wb + 0 * 4096, smem_base + (AOFF + 0 * 4096) * 2);
        tdm_load_chunk(wb + 1 * 4096, smem_base + (AOFF + 1 * 4096) * 2);
        __builtin_amdgcn_s_wait_tensorcnt(1);   // chunk 0 complete (in-order)
    }
    __syncthreads();     // window + chunk 0 visible to all waves
#endif

    v8f acc = {};
    for (int ci = 0; ci < 18; ++ci) {
        const int p   = ci >> 1;
        const int ic0 = (ci & 1) * 32;
        const int kh  = p / 3, kw = p % 3;
        const int buf = ci & 1;

#if !HAVE_TDM
        __syncthreads();
        {   // fallback: cooperative staging, hi+lo 8 KB = 512 x b128
            const uint4* s4 = (const uint4*)(wb + (size_t)ci * 4096);
            uint4* d4 = (uint4*)&smem[AOFF + buf * 4096];
            d4[tid]       = s4[tid];
            d4[tid + 256] = s4[tid + 256];
        }
        __syncthreads();
#endif
        const __bf16* Ab = &smem[AOFF + buf * 4096];

        // A fragment (16x32 bf16): lane -> M = lane%16; K slots per ISA table.
        v16bf ah, al, bh, bl;
        #pragma unroll
        for (int s = 0; s < 16; ++s) {
            const int kv = 2 * (s >> 1) + (s & 1)
                         + (hsel ? ((s < 8) ? 8 : 16) : ((s < 8) ? 0 : 8));
            ah[s] = Ab[kv * OUTC + oc0 + m];
            al[s] = Ab[2048 + kv * OUTC + oc0 + m];
        }
        // B fragment (32x16 bf16): lane -> N = lane%16; K slots per ISA table.
        const int bbase = ic0 * WROW + kh * WSTR + nOff + kw + m;
        #pragma unroll
        for (int s = 0; s < 16; ++s) {
            const int kv = 2 * (s >> 1) + (s & 1)
                         + (hsel ? ((s < 8) ? 8 : 16) : ((s < 8) ? 0 : 8));
            bh[s] = smem[XH + bbase + kv * WROW];
            bl[s] = smem[XL + bbase + kv * WROW];
        }
        // Split-precision product: hi*hi + hi*lo + lo*hi (~fp32 accuracy)
        acc = __builtin_amdgcn_wmma_f32_16x16x32_bf16(false, ah, false, bh,
                                                      (short)0, acc, false, false);
        acc = __builtin_amdgcn_wmma_f32_16x16x32_bf16(false, ah, false, bl,
                                                      (short)0, acc, false, false);
        acc = __builtin_amdgcn_wmma_f32_16x16x32_bf16(false, al, false, bh,
                                                      (short)0, acc, false, false);

#if HAVE_TDM
        __syncthreads();                 // all waves done reading buf
        if (wid == 0) {
            if (ci + 2 < 18) {
                tdm_load_chunk(wb + (size_t)(ci + 2) * 4096,
                               smem_base + (AOFF + buf * 4096) * 2);
                __builtin_amdgcn_s_wait_tensorcnt(1);   // chunk ci+1 done
            } else {
                __builtin_amdgcn_s_wait_tensorcnt(0);   // drain
            }
        }
        __syncthreads();                 // next buffer visible to all waves
#endif
    }

    // Epilogue: C layout lane%16 -> pixel, VGPR r -> oc = oc0 + r + 8*(lane/16).
    const int wcol = w0 + nOff + (lane & 15);
    #pragma unroll
    for (int r = 0; r < 8; ++r) {
        const int oc = oc0 + r + 8 * hsel;
        const float v = acc[r] + bias[b * OUTC + oc];
        out[(((size_t)b * OUTC + oc) * HWDIM + h) * HWDIM + wcol] = v;
    }
}

// ---------------------------------------------------------------------------
extern "C" void kernel_launch(void* const* d_in, const int* in_sizes, int n_in,
                              void* d_out, int out_size, void* d_ws, size_t ws_size,
                              hipStream_t stream) {
    (void)in_sizes; (void)n_in; (void)out_size; (void)ws_size;
    const float* x    = (const float*)d_in[0];   // (16,64,128,128)
    const float* cond = (const float*)d_in[1];   // (16,256)
    const float* Wc   = (const float*)d_in[2];   // (36928,256)
    const float* bc   = (const float*)d_in[3];   // (36928,)

    __bf16* w_pk = (__bf16*)d_ws;                      // 16*18*4096 bf16 = 2.25 MB
    float*  bias = (float*)((char*)d_ws + (size_t)NB * 18 * 4096 * 2);

    hyper_gemm<<<577, 128, 0, stream>>>(cond, Wc, bc, w_pk, bias);
    conv_wmma<<<dim3(512, NB), 256, 0, stream>>>(x, w_pk, bias, (float*)d_out);
}
